// CKAN_46866683133999
// MI455X (gfx1250) — matrix operand order repaired and checked by reading, not compiled, not run on hardware
//
#include <hip/hip_runtime.h>
#include <hip/hip_bf16.h>

#define NENT  100000
#define DIM   64
#define BATCH 4096
#define TT    64

typedef __attribute__((ext_vector_type(16))) __bf16 v16bf;
typedef __attribute__((ext_vector_type(8)))  float  v8f;

union FragB16 {
  v16bf v;
  uint4 q[2];
};

__device__ __forceinline__ unsigned short f2bf(float f) {
  unsigned int u = __float_as_uint(f);
  u += 0x7FFFu + ((u >> 16) & 1u);          // round-to-nearest-even
  return (unsigned short)(u >> 16);
}

// ---------------------------------------------------------------------------
// Pack W1 (128x64) and W2 (64x64) fp32 -> bf16 WMMA B-fragments in workspace.
// B-matrix (KxN = 32x16) wave32 layout: lane L holds n = nt*16 + (L&15),
// k = kt*32 + (L<16 ? 0 : 8) + (j&7) + (j>=8 ? 16 : 0) for j = 0..15.
// W1: 16 frags (kt=0..3, nt=0..3) * 32 lanes * 16 elems = 8192 ushorts
// W2:  8 frags (kt=0..1, nt=0..3)                      = 4096 ushorts (after W1)
// ---------------------------------------------------------------------------
__global__ void ckan_pack_w(const float* __restrict__ W1,
                            const float* __restrict__ W2,
                            unsigned short* __restrict__ wsW) {
  int e = blockIdx.x * blockDim.x + threadIdx.x;
  if (e < 8192) {
    int frag = e >> 9, rem = e & 511;
    int kt = frag >> 2, nt = frag & 3;
    int lane = rem >> 4, j = rem & 15;
    int k = kt * 32 + ((lane < 16) ? 0 : 8) + (j & 7) + ((j >> 3) << 4);
    int n = nt * 16 + (lane & 15);
    wsW[e] = f2bf(W1[k * DIM + n]);
  } else if (e < 8192 + 4096) {
    int e2 = e - 8192;
    int frag = e2 >> 9, rem = e2 & 511;
    int kt = frag >> 2, nt = frag & 3;
    int lane = rem >> 4, j = rem & 15;
    int k = kt * 32 + ((lane < 16) ? 0 : 8) + (j & 7) + ((j >> 3) << 4);
    int n = nt * 16 + (lane & 15);
    wsW[8192 + e2] = f2bf(W2[k * DIM + n]);
  }
}

// ---------------------------------------------------------------------------
// Fused knowledge-attention: grid (B, 4 calls), 128 threads (4 waves).
// Wave w owns M-tile w (16 of the T=64 rows); each wave does all 4 N-tiles.
// ---------------------------------------------------------------------------
__global__ __launch_bounds__(128)
void ckan_attention(const int* __restrict__ user_h, const int* __restrict__ user_r,
                    const int* __restrict__ user_t,
                    const int* __restrict__ item_h, const int* __restrict__ item_r,
                    const int* __restrict__ item_t,
                    const float* __restrict__ ent, const float* __restrict__ rel,
                    const unsigned short* __restrict__ wsW,
                    const float* __restrict__ W3,
                    float* __restrict__ attn) {
  __shared__ unsigned short xlds[64][128];  // 16 KB  concat(h,r) bf16
  __shared__ unsigned short a1lds[64][64];  //  8 KB  relu(x@W1) bf16
  __shared__ float a2f[64][65];             // ~16.6 KB relu(a1@W2) f32 (pad->no bank conflict)
  __shared__ float w3lds[64];
  __shared__ float plds[64];
  __shared__ float wlds[64];
  __shared__ int   tind[64];
  __shared__ float partial[64];

  const int b   = blockIdx.x;
  const int c   = blockIdx.y;        // 0,1: user layers; 2,3: item layers
  const int tid = threadIdx.x;
  const int lane = tid & 31;
  const int wid  = tid >> 5;

  const long ofs = (long)(c & 1) * BATCH * TT + (long)b * TT;
  const int* hp = ((c < 2) ? user_h : item_h) + ofs;
  const int* rp = ((c < 2) ? user_r : item_r) + ofs;
  const int* tp = ((c < 2) ? user_t : item_t) + ofs;

  // ---- stage tail-gather indices + W3 ----
  if (tid < 64) tind[tid] = tp[tid];
  else          w3lds[tid - 64] = W3[tid - 64];

  // ---- stage x = [h_emb | r_emb] -> bf16 LDS (64 rows x 32 float4) ----
  for (int idx = tid; idx < 64 * 32; idx += 128) {
    int t  = idx >> 5;
    int c4 = idx & 31;
    const float* src = (c4 < 16) ? (ent + (long)hp[t] * DIM + c4 * 4)
                                 : (rel + (long)rp[t] * DIM + (c4 - 16) * 4);
    float4 f = *(const float4*)src;
    unsigned int lo = ((unsigned int)f2bf(f.y) << 16) | f2bf(f.x);
    unsigned int hi = ((unsigned int)f2bf(f.w) << 16) | f2bf(f.z);
    *(uint2*)&xlds[t][c4 * 4] = make_uint2(lo, hi);
  }
  __syncthreads();

  // A-fragment addressing (16-bit A 16x32): lane L -> row L&15,
  // K chunks {kb..kb+7, kb+16..kb+23}, kb = (L<16)?0:8
  const int mrow = lane & 15;
  const int kb   = (lane < 16) ? 0 : 8;
  const int trow = (wid << 4) + mrow;
  const v8f vzero = (v8f)(0.0f);

  // ---- GEMM1: [64x128] x W1[128x64], K = 4 x 32 ----
  v8f acc[4] = {vzero, vzero, vzero, vzero};
  for (int kt = 0; kt < 4; ++kt) {
    FragB16 fa;
    int off = (kt << 5) + kb;
    fa.q[0] = *(const uint4*)&xlds[trow][off];
    fa.q[1] = *(const uint4*)&xlds[trow][off + 16];
#pragma unroll
    for (int nt = 0; nt < 4; ++nt) {
      FragB16 fb;
      const uint4* p = (const uint4*)wsW + ((kt * 4 + nt) * 64 + lane * 2);
      fb.q[0] = p[0]; fb.q[1] = p[1];
      acc[nt] = __builtin_amdgcn_wmma_f32_16x16x32_bf16(
          false, fa.v, false, fb.v, (short)0, acc[nt], false, false);
    }
  }

  // relu -> bf16 -> LDS.  D layout: lane holds n = lane&15, rows (lane<16? i : 8+i)
  const int nl = lane & 15;
  const int mb = (lane < 16) ? 0 : 8;
#pragma unroll
  for (int nt = 0; nt < 4; ++nt)
#pragma unroll
    for (int i = 0; i < 8; ++i) {
      float v = acc[nt][i];
      a1lds[(wid << 4) + mb + i][(nt << 4) + nl] = f2bf(v > 0.f ? v : 0.f);
    }
  __syncthreads();

  // ---- GEMM2: [64x64] x W2[64x64], K = 2 x 32 ----
  v8f bcc[4] = {vzero, vzero, vzero, vzero};
  for (int kt = 0; kt < 2; ++kt) {
    FragB16 fa;
    int off = (kt << 5) + kb;
    fa.q[0] = *(const uint4*)&a1lds[trow][off];
    fa.q[1] = *(const uint4*)&a1lds[trow][off + 16];
#pragma unroll
    for (int nt = 0; nt < 4; ++nt) {
      FragB16 fb;
      const uint4* p = (const uint4*)wsW + (1024 + (kt * 4 + nt) * 64 + lane * 2);
      fb.q[0] = p[0]; fb.q[1] = p[1];
      bcc[nt] = __builtin_amdgcn_wmma_f32_16x16x32_bf16(
          false, fa.v, false, fb.v, (short)0, bcc[nt], false, false);
    }
  }
#pragma unroll
  for (int nt = 0; nt < 4; ++nt)
#pragma unroll
    for (int i = 0; i < 8; ++i) {
      float v = bcc[nt][i];
      a2f[(wid << 4) + mb + i][(nt << 4) + nl] = v > 0.f ? v : 0.f;
    }
  __syncthreads();

  // ---- a3 = a2 @ W3 ; sigmoid ----
  if (tid < 64) {
    float s = 0.f;
    for (int n = 0; n < 64; ++n) s += a2f[tid][n] * w3lds[n];
    plds[tid] = 1.f / (1.f + __expf(-s));
  }
  __syncthreads();

  // ---- softmax over T=64 (redundant block-wide scan; LDS broadcast reads) ----
  float mx = -1e30f;
  for (int t = 0; t < 64; ++t) mx = fmaxf(mx, plds[t]);
  float sum = 0.f;
  for (int t = 0; t < 64; ++t) sum += __expf(plds[t] - mx);
  if (tid < 64) wlds[tid] = __expf(plds[tid] - mx) / sum;
  __syncthreads();

  // ---- out[d] = sum_t w[t] * ent[t_idx[t]][d]  (coalesced row reads, L2-hot) ----
  const int d    = tid & 63;
  const int half = tid >> 6;
  float o = 0.f;
  for (int t = half * 32; t < half * 32 + 32; ++t)
    o += wlds[t] * ent[(long)tind[t] * DIM + d];
  if (half) partial[d] = o;
  __syncthreads();
  if (!half)
    attn[((long)c * BATCH + b) * DIM + d] = o + partial[d];
}

// ---------------------------------------------------------------------------
// Finalize: e_u/e_v assembly + dot + sigmoid. One 64-thread block per b.
// ---------------------------------------------------------------------------
__global__ __launch_bounds__(64)
void ckan_finalize(const int* __restrict__ items,
                   const int* __restrict__ user_h, const int* __restrict__ item_h,
                   const float* __restrict__ ent,
                   const float* __restrict__ attn,
                   float* __restrict__ out) {
  __shared__ int uh[64], ih[64];
  __shared__ float red[2];
  const int b = blockIdx.x;
  const int d = threadIdx.x;

  uh[d] = user_h[(long)b * TT + d];   // layer-0 slice
  ih[d] = item_h[(long)b * TT + d];
  __syncthreads();

  float su = 0.f, si = 0.f;
  for (int t = 0; t < 64; ++t) {
    su += ent[(long)uh[t] * DIM + d];
    si += ent[(long)ih[t] * DIM + d];
  }
  const float inv = 1.f / 64.f;
  const long bd = (long)b * DIM + d;
  const long cstride = (long)BATCH * DIM;
  float eu = su * inv + attn[0 * cstride + bd] + attn[1 * cstride + bd];
  float ev = ent[(long)items[b] * DIM + d]
           + attn[2 * cstride + bd] + attn[3 * cstride + bd] + si * inv;

  float v = eu * ev;
  for (int off = 16; off > 0; off >>= 1) v += __shfl_xor(v, off, 32);
  if ((d & 31) == 0) red[d >> 5] = v;
  __syncthreads();
  if (d == 0) out[b] = 1.f / (1.f + __expf(-(red[0] + red[1])));
}

// ---------------------------------------------------------------------------
extern "C" void kernel_launch(void* const* d_in, const int* in_sizes, int n_in,
                              void* d_out, int out_size, void* d_ws, size_t ws_size,
                              hipStream_t stream) {
  const int*   items  = (const int*)d_in[0];
  const int*   user_h = (const int*)d_in[1];
  const int*   user_r = (const int*)d_in[2];
  const int*   user_t = (const int*)d_in[3];
  const int*   item_h = (const int*)d_in[4];
  const int*   item_r = (const int*)d_in[5];
  const int*   item_t = (const int*)d_in[6];
  const float* ent    = (const float*)d_in[7];
  const float* rel    = (const float*)d_in[8];
  const float* W1     = (const float*)d_in[9];
  const float* W2     = (const float*)d_in[10];
  const float* W3     = (const float*)d_in[11];
  float* out = (float*)d_out;

  unsigned short* wsW = (unsigned short*)d_ws;          // 12288 ushorts = 24576 B
  float* attn = (float*)((char*)d_ws + 24576);          // 4 * B * 64 floats = 4 MB

  ckan_pack_w<<<dim3(48), dim3(256), 0, stream>>>(W1, W2, wsW);
  ckan_attention<<<dim3(BATCH, 4), dim3(128), 0, stream>>>(
      user_h, user_r, user_t, item_h, item_r, item_t, ent, rel, wsW, W3, attn);
  ckan_finalize<<<dim3(BATCH), dim3(64), 0, stream>>>(items, user_h, item_h, ent, attn, out);
}
